// OIMLoss_siameseoffline_43001212567997
// MI455X (gfx1250) — matrix operand order repaired and checked by reading, not compiled, not run on hardware
//
#include <hip/hip_runtime.h>

// ---------------- problem constants (from reference) ----------------
#define B_   256
#define N_   131072
#define D_   2048
#define H_   512
#define TOPK 500
#define NEG_SLOPE 0.01f
#define MASK_VAL  (-10000.0f)

typedef __attribute__((ext_vector_type(16))) __bf16 v16bf;
typedef __attribute__((ext_vector_type(8)))  __bf16 v8bf;
typedef __attribute__((ext_vector_type(8)))  float  v8f;
typedef __attribute__((ext_vector_type(4)))  unsigned v4u;
typedef __attribute__((ext_vector_type(8)))  int      v8i;
typedef __attribute__((ext_vector_type(4)))  int      v4i;

__device__ __forceinline__ unsigned hash_u32(unsigned x) {
    x ^= x >> 16; x *= 0x7feb352du;
    x ^= x >> 15; x *= 0x846ca68bu;
    x ^= x >> 16; return x;
}

// float -> uint key such that ascending key == DESCENDING float value.
__device__ __forceinline__ unsigned dkey(float f) {
    unsigned u = __float_as_uint(f);
    unsigned flip = (unsigned)(-(int)(u >> 31)) | 0x80000000u;
    return ~(u ^ flip);
}

__device__ __forceinline__ float softplus_f(float x) {
    if (x > 20.0f)  return x;
    if (x < -20.0f) return __expf(x);
    return log1pf(__expf(x));
}

// A-fragment loader: 16-bit A 16x32, contiguous k order (k = 16*half + s).
__device__ __forceinline__ v16bf load_a_frag(const __bf16* rowp, int k0, int half) {
    union { v16bf v; v8bf h[2]; } u;
    const __bf16* p = rowp + k0 + half * 16;
    u.h[0] = *(const v8bf*)(p);
    u.h[1] = *(const v8bf*)(p + 8);
    return u.v;
}

// ---------------- 1) row-normalize inputs, emit f32 + bf16 ----------------
__global__ void k_normalize(const float* __restrict__ in,
                            float* __restrict__ xn, __bf16* __restrict__ xb) {
    const int row = blockIdx.x;
    const float* r = in + (size_t)row * D_;
    float s = 0.f;
    for (int i = threadIdx.x; i < D_; i += 256) { float v = r[i]; s += v * v; }
    __shared__ float red[256];
    red[threadIdx.x] = s; __syncthreads();
    for (int o = 128; o > 0; o >>= 1) {
        if (threadIdx.x < o) red[threadIdx.x] += red[threadIdx.x + o];
        __syncthreads();
    }
    const float inv = rsqrtf(red[0] + 1e-30f);
    for (int i = threadIdx.x; i < D_; i += 256) {
        float v = r[i] * inv;
        xn[(size_t)row * D_ + i] = v;
        xb[(size_t)row * D_ + i] = (__bf16)v;
    }
}

// ---------------- 2) sims GEMM: x_bf16 [256xD] @ SF^T, masked store ----------------
#define MT_PER_WAVE 8
__global__ void k_sims_wmma(const __bf16* __restrict__ xb,
                            const float*  __restrict__ sf,
                            const int*    __restrict__ labels,
                            const int*    __restrict__ targets,
                            float* __restrict__ sims) {
    __shared__ int s_tgt[B_];
    if (threadIdx.x < B_) s_tgt[threadIdx.x] = targets[threadIdx.x];
    __syncthreads();

    const int wave  = threadIdx.x >> 5;
    const int lane  = threadIdx.x & 31;
    const int half  = lane >> 4;
    const int l     = lane & 15;
    const int nTile = blockIdx.x * 8 + wave;          // 0..8191
    const int mBase = blockIdx.y * MT_PER_WAVE;       // tile units

    const int col = (nTile << 4) + l;
    const float* bcol = sf + (size_t)col * D_;

    v8f acc[MT_PER_WAVE];
#pragma unroll
    for (int mt = 0; mt < MT_PER_WAVE; ++mt) acc[mt] = (v8f)(0.f);

    for (int k0 = 0; k0 < D_; k0 += 32) {
        const float* bp = bcol + k0 + (half ? 16 : 0);
        __builtin_prefetch((const void*)(bp + 64), 0, 0);
        v8f b0 = *(const v8f*)(bp);
        v8f b1 = *(const v8f*)(bp + 8);
        v16bf bb;
#pragma unroll
        for (int j = 0; j < 8; ++j) { bb[j] = (__bf16)b0[j]; bb[8 + j] = (__bf16)b1[j]; }

#pragma unroll
        for (int mt = 0; mt < MT_PER_WAVE; ++mt) {
            const int mrow = ((mBase + mt) << 4) + l;
            v16bf aa = load_a_frag(xb + (size_t)mrow * D_, k0, half);
            acc[mt] = __builtin_amdgcn_wmma_f32_16x16x32_bf16(
                false, aa, false, bb, (short)0, acc[mt], false, false);
        }
    }

    const int lab = labels[col];
#pragma unroll
    for (int mt = 0; mt < MT_PER_WAVE; ++mt) {
#pragma unroll
        for (int r = 0; r < 8; ++r) {
            const int m = ((mBase + mt) << 4) + (half ? 8 : 0) + r;
            float v = acc[mt][r];
            if (lab == s_tgt[m]) v = MASK_VAL;
            sims[(size_t)m * N_ + col] = v;
        }
    }
}

// ---------------- 3) per-row rank-r radix select (the r-th largest) ----------------
__global__ void k_negselect(const float* __restrict__ sims, int* __restrict__ neg_sel) {
    const int row = blockIdx.x, tid = threadIdx.x;
    const float* rowp = sims + (size_t)row * N_;
    __shared__ int      hist[256];
    __shared__ unsigned s_prefix;
    __shared__ int      s_rr;

    const int r0 = (int)(hash_u32(0x9E3779B9u * (unsigned)row + 42u) % TOPK);
    if (tid == 0) { s_prefix = 0u; s_rr = r0; }
    __syncthreads();

#pragma unroll
    for (int pass = 0; pass < 4; ++pass) {
        const int shift = 24 - pass * 8;
        const unsigned mask =
            (pass == 0) ? 0x00000000u :
            (pass == 1) ? 0xFF000000u :
            (pass == 2) ? 0xFFFF0000u : 0xFFFFFF00u;
        hist[tid] = 0; __syncthreads();
        const unsigned pref = s_prefix;
        for (int i = tid; i < N_; i += 256) {
            unsigned k = dkey(rowp[i]);
            if (((k ^ pref) & mask) == 0u)
                atomicAdd(&hist[(int)((k >> shift) & 255u)], 1);
        }
        __syncthreads();
        if (tid == 0) {
            int rr = s_rr;
            int b = 255;
            for (int j = 0; j < 255; ++j) {
                const int h = hist[j];
                if (rr < h) { b = j; break; }
                rr -= h;
            }
            s_rr = rr;
            s_prefix = pref | ((unsigned)b << shift);
        }
        __syncthreads();
    }

    const unsigned key = s_prefix;
    const int rr = s_rr;
    __shared__ int cnt[256];
    const int CH = N_ / 256;
    int c = 0;
    const int base = tid * CH;
    for (int i = 0; i < CH; ++i) if (dkey(rowp[base + i]) == key) ++c;
    cnt[tid] = c; __syncthreads();
    if (tid == 0) {
        int t = rr;
        int ch = 255;
        for (int j = 0; j < 255; ++j) {
            const int h = cnt[j];
            if (t < h) { ch = j; break; }
            t -= h;
        }
        int idx = 0;
        const int bb = ch * CH;
        for (int i = 0; i < CH; ++i) {
            if (dkey(rowp[bb + i]) == key) {
                if (t == 0) { idx = bb + i; break; }
                --t;
            }
        }
        neg_sel[row] = idx;
    }
}

// ---------------- 4) uniform-random positive via hash-argmax ----------------
__global__ void k_posselect(const int* __restrict__ labels, const int* __restrict__ targets,
                            int* __restrict__ pos_sel) {
    const int row = blockIdx.x, tid = threadIdx.x;
    const int tgt = targets[row];
    unsigned bestk = 0u; int besti = N_;
    for (int i = tid; i < N_; i += 256) {
        if (labels[i] == tgt) {
            unsigned k = hash_u32(hash_u32((unsigned)i * 0x9E3779B9u) ^
                                  ((unsigned)row * 0x85EBCA6Bu + 777u)) | 1u;
            if (k > bestk || (k == bestk && i < besti)) { bestk = k; besti = i; }
        }
    }
    __shared__ unsigned sk[256]; __shared__ int si[256];
    sk[tid] = bestk; si[tid] = besti; __syncthreads();
    for (int o = 128; o > 0; o >>= 1) {
        if (tid < o) {
            if (sk[tid + o] > sk[tid] || (sk[tid + o] == sk[tid] && si[tid + o] < si[tid])) {
                sk[tid] = sk[tid + o]; si[tid] = si[tid + o];
            }
        }
        __syncthreads();
    }
    if (tid == 0) pos_sel[row] = (si[0] >= N_) ? 0 : si[0];
}

// ---------------- 5) U = |x_norm - SF[sel]| as bf16 ----------------
__global__ void k_absdiff(const float* __restrict__ xn, const float* __restrict__ sf,
                          const int* __restrict__ pos_sel, const int* __restrict__ neg_sel,
                          __bf16* __restrict__ U) {
    const int row = blockIdx.x;
    const int b = row & (B_ - 1);
    const int sel = (row < B_) ? pos_sel[b] : neg_sel[b];
    const float* xr = xn + (size_t)b * D_;
    const float* fr = sf + (size_t)sel * D_;
    for (int i = threadIdx.x; i < D_; i += 256)
        U[(size_t)row * D_ + i] = (__bf16)fabsf(xr[i] - fr[i]);
}

// ---------------- 6) head GEMM: H = leakyrelu(U @ fc1_w + b1) ----------------
// B-tile (fc1_w, stride H_=512 f32) is a strided 2D gather -> stage via Tensor
// Data Mover into per-wave LDS double buffers; A loaded direct; WMMA bf16.
#if defined(__AMDGCN__) && __has_builtin(__builtin_amdgcn_tensor_load_to_lds)
#define HEAD_USE_TDM 1
#else
#define HEAD_USE_TDM 0
#endif

__global__ void k_head_wmma(const __bf16* __restrict__ U, const float* __restrict__ fc1w,
                            const float* __restrict__ fc1b, __bf16* __restrict__ HB) {
    const int wave = threadIdx.x >> 5;
    const int lane = threadIdx.x & 31;
    const int half = lane >> 4;
    const int l    = lane & 15;
    const int wid  = blockIdx.x * 8 + wave;   // 0..1023
    const int mt   = wid >> 5;                // 0..31 (M tile)
    const int nt   = wid & 31;                // 0..31 (N tile)

    const int col  = (nt << 4) + l;           // output neuron
    const int mrow = (mt << 4) + l;

    v8f acc = (v8f)(0.f);

#if HEAD_USE_TDM
    // per-wave double-buffered staging: 32(K) x 16(N) f32 tile, row-major K x N
    __shared__ float bstage[8][2][32 * 16];

    // TDM descriptor builder (ISA 8.3/8.4): 2D tensor, groups 2/3 unused.
    auto tdm_issue = [&](int buf, int k0) {
        const unsigned lds_off =
            (unsigned)(unsigned long long)(const void*)&bstage[wave][buf][0];
        const unsigned long long ga =
            (unsigned long long)(const void*)(fc1w + (size_t)k0 * H_ + (nt << 4));
        v4u g0;
        g0[0] = 1u;                                        // count=1, user mode
        g0[1] = lds_off;                                   // lds_addr (bytes)
        g0[2] = (unsigned)(ga & 0xFFFFFFFFull);            // global_addr[31:0]
        g0[3] = (unsigned)((ga >> 32) & 0x1FFFFFFull)      // global_addr[56:32]
                | (2u << 30);                              // type=2 ("image")
        v8i g1;
        g1[0] = (int)(2u << 16);                           // data_size=2 -> 4 bytes
        g1[1] = (int)(((unsigned)H_ & 0xFFFFu) << 16);     // tensor_dim0[15:0]=512
        g1[2] = (int)(((unsigned)D_ & 0xFFFFu) << 16);     // dim0 hi=0 | tensor_dim1[15:0]=2048
        g1[3] = (int)(16u << 16);                          // dim1 hi=0 | tile_dim0=16
        g1[4] = (int)32u;                                  // tile_dim1=32, tile_dim2=0
        g1[5] = (int)(unsigned)H_;                         // tensor_dim0_stride=512
        g1[6] = 0;                                         // stride0 hi | dim1_stride lo
        g1[7] = 0;
        v4i z4 = {0, 0, 0, 0};
        v8i z8 = {0, 0, 0, 0, 0, 0, 0, 0};
        __builtin_amdgcn_tensor_load_to_lds(g0, g1, z4, z4, z8, 0);
    };

    int buf = 0;
    tdm_issue(0, 0);
    for (int k0 = 0; k0 < D_; k0 += 32) {
        if (k0 + 32 < D_) {
            tdm_issue(buf ^ 1, k0 + 32);
            __builtin_amdgcn_s_wait_tensorcnt(1);   // current buffer complete
        } else {
            __builtin_amdgcn_s_wait_tensorcnt(0);
        }
        const float* bs = &bstage[wave][buf][0];
        v16bf bb;
#pragma unroll
        for (int j = 0; j < 16; ++j)
            bb[j] = (__bf16)bs[(half * 16 + j) * 16 + l];
        v16bf aa = load_a_frag(U + (size_t)mrow * D_, k0, half);
        acc = __builtin_amdgcn_wmma_f32_16x16x32_bf16(
            false, aa, false, bb, (short)0, acc, false, false);
        buf ^= 1;
    }
#else
    for (int k0 = 0; k0 < D_; k0 += 32) {
        v16bf aa = load_a_frag(U + (size_t)mrow * D_, k0, half);
        const float* wp = fc1w + (size_t)(k0 + (half ? 16 : 0)) * H_ + col;
        v16bf bb;
#pragma unroll
        for (int j = 0; j < 16; ++j) bb[j] = (__bf16)wp[(size_t)j * H_];
        acc = __builtin_amdgcn_wmma_f32_16x16x32_bf16(
            false, aa, false, bb, (short)0, acc, false, false);
    }
#endif

    const float bias = fc1b[col];
#pragma unroll
    for (int r = 0; r < 8; ++r) {
        const int m = (mt << 4) + (half ? 8 : 0) + r;
        float v = acc[r] + bias;
        v = (v > 0.f) ? v : v * NEG_SLOPE;    // LeakyReLU
        HB[(size_t)m * H_ + col] = (__bf16)v;
    }
}

// ---------------- 7) logits: 512-dot per row vs fc2_w ----------------
__global__ void k_logits(const __bf16* __restrict__ HB, const float* __restrict__ fc2w,
                         const float* __restrict__ fc2b, float* __restrict__ logits) {
    const int row = blockIdx.x, tid = threadIdx.x;
    float s = 0.f;
    for (int i = tid; i < H_; i += 256) s += (float)HB[(size_t)row * H_ + i] * fc2w[i];
    __shared__ float red[256];
    red[tid] = s; __syncthreads();
    for (int o = 128; o > 0; o >>= 1) {
        if (tid < o) red[tid] += red[tid + o];
        __syncthreads();
    }
    if (tid == 0) logits[row] = red[0] + fc2b[0];
}

// ---------------- 8) BCE-with-logits loss ----------------
__global__ void k_loss(const float* __restrict__ logits, float* __restrict__ out) {
    const int tid = threadIdx.x;
    const float x = logits[tid];
    const float v = (tid < B_) ? softplus_f(-x) : softplus_f(x);
    __shared__ float red[512];
    red[tid] = v; __syncthreads();
    for (int o = 256; o > 0; o >>= 1) {
        if (tid < o) red[tid] += red[tid + o];
        __syncthreads();
    }
    if (tid == 0) out[0] = red[0] * (1.0f / (float)B_);
}

// ---------------- launch ----------------
extern "C" void kernel_launch(void* const* d_in, const int* in_sizes, int n_in,
                              void* d_out, int out_size, void* d_ws, size_t ws_size,
                              hipStream_t stream) {
    const float* inputs  = (const float*)d_in[0];
    const int*   targets = (const int*)  d_in[1];
    const float* sf      = (const float*)d_in[2];
    const int*   labels  = (const int*)  d_in[3];
    const float* fc1w    = (const float*)d_in[4];
    const float* fc1b    = (const float*)d_in[5];
    const float* fc2w    = (const float*)d_in[6];
    const float* fc2b    = (const float*)d_in[7];
    float* out = (float*)d_out;

    char* w = (char*)d_ws;
    size_t off = 0;
    auto take = [&](size_t bytes) -> char* {
        char* p = w + off;
        off = (off + bytes + 255) & ~(size_t)255;
        return p;
    };
    float*  xn      = (float*) take((size_t)B_ * D_ * 4);
    __bf16* xb      = (__bf16*)take((size_t)B_ * D_ * 2);
    float*  sims    = (float*) take((size_t)B_ * N_ * 4);
    int*    neg_sel = (int*)   take((size_t)B_ * 4);
    int*    pos_sel = (int*)   take((size_t)B_ * 4);
    __bf16* U       = (__bf16*)take((size_t)2 * B_ * D_ * 2);
    __bf16* HB      = (__bf16*)take((size_t)2 * B_ * H_ * 2);
    float*  logits  = (float*) take((size_t)2 * B_ * 4);
    (void)ws_size; (void)n_in; (void)in_sizes; (void)out_size;

    k_normalize<<<B_, 256, 0, stream>>>(inputs, xn, xb);
    k_sims_wmma<<<dim3(N_ / 16 / 8, (B_ / 16) / MT_PER_WAVE), 256, 0, stream>>>(
        xb, sf, labels, targets, sims);
    k_negselect<<<B_, 256, 0, stream>>>(sims, neg_sel);
    k_posselect<<<B_, 256, 0, stream>>>(labels, targets, pos_sel);
    k_absdiff<<<2 * B_, 256, 0, stream>>>(xn, sf, pos_sel, neg_sel, U);
    k_head_wmma<<<(32 * 32) / 8, 256, 0, stream>>>(U, fc1w, fc1b, HB);
    k_logits<<<2 * B_, 256, 0, stream>>>(HB, fc2w, fc2b, logits);
    k_loss<<<1, 512, 0, stream>>>(logits, out);
}
